// LowRankAttention_77043123355929
// MI455X (gfx1250) — matrix operand order repaired and checked by reading, not compiled, not run on hardware
//
#include <hip/hip_runtime.h>
#include <hip/hip_bf16.h>

// ---------------------------------------------------------------------------
// Problem constants (from reference: B=4, L=4096, C=1024, H=8, d=64, r=512)
// ---------------------------------------------------------------------------
#define BB 4
#define LL 4096
#define CC 1024
#define HH 8
#define DD 64      // d
#define D2 128     // 2d
#define RR 512     // r = L^0.75
#define MT (BB*RR) // 2048 total low-rank tokens

#define KCH 256        // GEMM K-chunk staged in LDS
#define WROW 264       // padded LDS row stride (bf16 elems): 264*2B=528B -> bank shift 4/row

typedef __bf16 v16bf __attribute__((ext_vector_type(16)));
typedef float  v8f   __attribute__((ext_vector_type(8)));
typedef int    v4i   __attribute__((ext_vector_type(4)));
typedef int    v8i   __attribute__((ext_vector_type(8)));

union FragU { uint4 u[2]; v16bf v; };

// A-matrix fragment (16x32 bf16): lane l -> row m=l&15, kbase=16*(l>>4)
// elements e = A[m][k0+kbase+e], e = 0..15 (16 contiguous)
__device__ __forceinline__ v16bf load_afrag(const __bf16* __restrict__ A, int lda,
                                            int m, int k0, int lane) {
    int kb = k0 + ((lane >> 4) << 4);
    const __bf16* p = A + (long)m * lda + kb;
    FragU t;
    t.u[0] = *reinterpret_cast<const uint4*>(p);
    t.u[1] = *reinterpret_cast<const uint4*>(p + 8);
    return t.v;
}

// B-matrix fragment (32x16 bf16, B[k][n] = W[n][k]): lane l -> col n=l&15,
// elements e: k = k0 + 16*(l>>4) + e  (16 contiguous bf16 from row n of W)
__device__ __forceinline__ v16bf load_bfrag(const __bf16* __restrict__ W, int ldw,
                                            int n, int k0, int lane) {
    int kk = k0 + ((lane >> 4) << 4);
    const __bf16* p = W + (long)n * ldw + kk;
    FragU t;
    t.u[0] = *reinterpret_cast<const uint4*>(p);
    t.u[1] = *reinterpret_cast<const uint4*>(p + 8);
    return t.v;
}

// B-matrix fragment sourced from an LDS tile (row-major, WROW stride)
__device__ __forceinline__ v16bf lds_bfrag(const __bf16* base, int row, int kr, int lane) {
    const __bf16* p = base + row * WROW + kr + ((lane >> 4) << 4);
    FragU t;
    t.u[0] = *reinterpret_cast<const uint4*>(p);
    t.u[1] = *reinterpret_cast<const uint4*>(p + 8);
    return t.v;
}

__device__ __forceinline__ v8f wmma_bf16(v16bf a, v16bf b, v8f c) {
    return __builtin_amdgcn_wmma_f32_16x16x32_bf16(false, a, false, b, (short)0, c,
                                                   false, false);
}

// async global -> LDS copy of 16 bytes (ASYNCcnt-tracked, CDNA5)
typedef __attribute__((address_space(3))) __bf16 lds_bf16_t;
__device__ __forceinline__ void async_g2l_b128(const __bf16* g, __bf16* l) {
    unsigned lofs = (unsigned)(unsigned long)(lds_bf16_t*)l;   // 32-bit LDS offset
    asm volatile("global_load_async_to_lds_b128 %0, %1, off"
                 :: "v"(lofs), "v"(g) : "memory");
}
__device__ __forceinline__ void wait_async0() {
    asm volatile("s_wait_asynccnt 0x0" ::: "memory");
}

// TDM: DMA a 2D tile (rows x KCH bf16, row stride = ldw elems) from global
// into LDS at ldsOff, hardware-padding each 512B row with 16B (-> WROW stride).
// Issued by one wave; TENSORcnt-tracked.
__device__ __forceinline__ void tdm_stage_tile(const __bf16* g, unsigned ldsOff,
                                               int ldw, int nrows, int totRows) {
    unsigned long ga = (unsigned long)(const void*)g;
    v4i g0;
    g0[0] = 1;                                        // count=1 valid descriptor
    g0[1] = (int)ldsOff;                              // lds_addr (bytes)
    g0[2] = (int)(ga & 0xFFFFFFFFu);                  // global_addr[31:0]
    g0[3] = (int)((ga >> 32) & 0x01FFFFFFu) | (int)(2u << 30);  // addr[56:32] | type=2
    v8i g1;
    // data_size=1 (2B) | pad_enable | pad_interval=6 (128 DW = 512B) | pad_amount=3 (4 DW = 16B)
    g1[0] = (1 << 16) | (1 << 20) | (6 << 22) | (3 << 25);
    g1[1] = (ldw & 0xFFFF) << 16;                     // tensor_dim0[15:0]
    g1[2] = ((unsigned)ldw >> 16) | ((unsigned)(totRows & 0xFFFF) << 16); // dim0 hi | dim1 lo
    g1[3] = ((unsigned)totRows >> 16) | ((unsigned)KCH << 16);            // dim1 hi | tile_dim0
    g1[4] = nrows;                                    // tile_dim1 | tile_dim2=0
    g1[5] = ldw;                                      // tensor_dim0_stride[31:0]
    g1[6] = 0;
    g1[7] = 0;
    asm volatile("tensor_load_to_lds %0, %1" :: "s"(g0), "s"(g1) : "memory");
}

// ---------------------------------------------------------------------------
// float -> bf16 conversion (weights)
// ---------------------------------------------------------------------------
__global__ void f2bf_kernel(const float* __restrict__ s, __bf16* __restrict__ d, long n) {
    long i = (long)blockIdx.x * blockDim.x + threadIdx.x;
    if (i < n) d[i] = (__bf16)s[i];
}

// ---------------------------------------------------------------------------
// linterp downsample L=4096 -> R=512 and cast to bf16
// ---------------------------------------------------------------------------
__global__ void downsample_kernel(const float* __restrict__ x, __bf16* __restrict__ xd) {
    long idx = (long)blockIdx.x * blockDim.x + threadIdx.x;
    if (idx >= (long)MT * CC) return;
    int  c  = idx & (CC - 1);
    long bi = idx >> 10;
    int  i  = bi & (RR - 1);
    int  b  = (int)(bi >> 9);
    float coord = (i + 0.5f) * ((float)LL / (float)RR) - 0.5f;
    coord = fminf(fmaxf(coord, 0.0f), (float)(LL - 1));
    int lo = (int)floorf(coord);
    int hi = lo + 1 < LL ? lo + 1 : LL - 1;
    float w = coord - (float)lo;
    const float* xb = x + (long)b * LL * CC + c;
    float v = xb[(long)lo * CC] * (1.0f - w) + xb[(long)hi * CC] * w;
    xd[idx] = (__bf16)v;
}

// ---------------------------------------------------------------------------
// lambda scalar = exp(sum lq1*lk1) - exp(sum lq2*lk2) + 0.8
// ---------------------------------------------------------------------------
__global__ __launch_bounds__(64) void lambda_kernel(const float* __restrict__ lq1,
                                                    const float* __restrict__ lk1,
                                                    const float* __restrict__ lq2,
                                                    const float* __restrict__ lk2,
                                                    float* __restrict__ lam) {
    __shared__ float s1[64], s2[64];
    int t = threadIdx.x;
    s1[t] = lq1[t] * lk1[t];
    s2[t] = lq2[t] * lk2[t];
    __syncthreads();
    for (int st = 32; st > 0; st >>= 1) {
        if (t < st) { s1[t] += s1[t + st]; s2[t] += s2[t + st]; }
        __syncthreads();
    }
    if (t == 0) *lam = expf(s1[0]) - expf(s2[0]) + 0.8f;
}

// ---------------------------------------------------------------------------
// Generic bf16 GEMM:  out[m,n] = sum_k A[m,k] * W[n,k]   (f32 accumulate)
// Block = 128 threads (4 waves). Tile 64(M) x 64(N); wave w owns rows w*16..+16.
// W-tile (64 x KCH) staged into LDS either by the Tensor Data Mover
// (USE_TDM=1, one tensor_load_to_lds per chunk, TENSORcnt) or by per-lane
// async global->LDS copies (USE_TDM=0, ASYNCcnt). B fragments read from LDS;
// A fragments direct from global (per-wave unique, L2-resident).
// EPI: 0 = plain f32 store, 1 = SiLU then store.
// ---------------------------------------------------------------------------
template <int EPI, int USE_TDM>
__global__ __launch_bounds__(128) void gemm_bf16_kernel(
        const __bf16* __restrict__ A, const __bf16* __restrict__ W,
        float* __restrict__ out, int M, int N, int K,
        long aStride, long wStride,
        int innerCount, long oInnerStride, long oOuterStride, int outLd) {
    __shared__ __bf16 wt[64 * WROW];   // ~33 KB

    int z = blockIdx.z;
    A   += (long)z * aStride;
    W   += (long)z * wStride;
    out += (long)(z / innerCount) * oOuterStride + (long)(z % innerCount) * oInnerStride;

    int lane = threadIdx.x & 31;
    int wv   = threadIdx.x >> 5;
    int m0   = blockIdx.x * 64 + wv * 16;
    int n0   = blockIdx.y * 64;
    int mA   = m0 + (lane & 15);

    v8f acc[4] = {};
    for (int kch = 0; kch < K; kch += KCH) {
        // stage W[n0..n0+63][kch..kch+KCH) into LDS
        if (USE_TDM) {
            if (threadIdx.x < 32) {   // wave 0 drives the DMA
                unsigned ldsOff = (unsigned)(unsigned long)(lds_bf16_t*)&wt[0];
                tdm_stage_tile(W + (long)n0 * K + kch, ldsOff, K, 64, N);
                __builtin_amdgcn_s_wait_tensorcnt(0);
            }
        } else {
            for (int i = threadIdx.x; i < (64 * KCH) / 8; i += 128) {
                int e   = i << 3;
                int row = e >> 8;       // KCH = 256 elems per row
                int col = e & (KCH - 1);
                async_g2l_b128(W + (long)(n0 + row) * K + kch + col,
                               &wt[row * WROW + col]);
            }
            wait_async0();
        }
        __syncthreads();

#pragma unroll 2
        for (int kr = 0; kr < KCH; kr += 32) {
            v16bf a = load_afrag(A, K, mA, kch + kr, lane);
            v16bf bfr[4];
#pragma unroll
            for (int nt = 0; nt < 4; ++nt)
                bfr[nt] = lds_bfrag(wt, nt * 16 + (lane & 15), kr, lane);
#pragma unroll
            for (int nt = 0; nt < 4; ++nt)
                acc[nt] = wmma_bf16(a, bfr[nt], acc[nt]);
            if (kch + kr + 32 < K)
                __builtin_prefetch(A + (long)mA * K + kch + kr + 32, 0, 1);
        }
        __syncthreads();
    }

    int mr = m0 + ((lane >> 4) << 3);
    int nc = lane & 15;
#pragma unroll
    for (int nt = 0; nt < 4; ++nt) {
#pragma unroll
        for (int j = 0; j < 8; ++j) {
            float v = acc[nt][j];
            if (EPI == 1) v = v / (1.0f + __expf(-v));   // SiLU
            out[(long)(mr + j) * outLd + n0 + nt * 16 + nc] = v;
        }
    }
}

// ---------------------------------------------------------------------------
// Per-(b,i,h,s) rmsnorm over d=64 + RoPE.  NOTE: reference applies rope on
// shape (B,r,H,d) so positions come from the HEAD axis (pos = h), not i.
// dst layout: [(b*H+h)*R + i]*D + j  (bf16)
// ---------------------------------------------------------------------------
__global__ __launch_bounds__(64) void qknorm_rope_kernel(
        const float* __restrict__ Src, __bf16* __restrict__ dst1,
        __bf16* __restrict__ dst2, const float* __restrict__ w) {
    __shared__ float sh[64];
    __shared__ float xs[64];
    int j   = threadIdx.x;
    int blk = blockIdx.x;
    int s      = blk & 1;
    int h      = (blk >> 1) & (HH - 1);
    int rowIdx = blk >> 4;          // b*R + i
    int i      = rowIdx & (RR - 1);
    int b      = rowIdx >> 9;

    float val = Src[(long)rowIdx * CC + h * D2 + s * DD + j];
    sh[j] = val * val;
    __syncthreads();
    for (int st = 32; st > 0; st >>= 1) {
        if (j < st) sh[j] += sh[j + st];
        __syncthreads();
    }
    float rms = sqrtf(sh[0] * (1.0f / (float)DD) + 1e-6f);
    float xn  = val / rms * w[j];
    xs[j] = xn;
    __syncthreads();

    int f = j & 31;
    float ang = (float)h * powf(10000.0f, -(float)f * (1.0f / 32.0f));
    float cs = cosf(ang), sn = sinf(ang);
    float o;
    if (j < 32) o = xn * cs - xs[j + 32] * sn;
    else        o = xs[j - 32] * sn + xn * cs;

    __bf16* d = (s == 0) ? dst1 : dst2;
    d[((long)(b * HH + h) * RR + i) * DD + j] = (__bf16)o;
}

// ---------------------------------------------------------------------------
// V rearrange + transpose:  Vt[(b*H+h)*D2 + j][i] = V[(b*R+i)*C + h*D2 + j]
// ---------------------------------------------------------------------------
__global__ void vtrans_kernel(const float* __restrict__ V, __bf16* __restrict__ Vt) {
    long idx = (long)blockIdx.x * blockDim.x + threadIdx.x;
    if (idx >= (long)BB * HH * D2 * RR) return;
    int  i   = idx & (RR - 1);
    long rem = idx >> 9;
    int  j   = rem & (D2 - 1);
    long bh  = rem >> 7;
    int  b   = (int)(bh >> 3);
    int  h   = (int)(bh & 7);
    Vt[idx] = (__bf16)V[((long)b * RR + i) * CC + h * D2 + j];
}

// ---------------------------------------------------------------------------
// Attention scores + dual softmax + combine:
// per block: one (b,h) and a 16-query tile; waves compute 16x512 scores via
// WMMA into LDS, block-wide softmax, P = p1 - lam*p2 stored bf16.
// Two phases (sub-head 1, then 2) to keep static LDS ~33 KB.
// ---------------------------------------------------------------------------
__global__ __launch_bounds__(128) void attn_kernel(
        const __bf16* __restrict__ q1r, const __bf16* __restrict__ q2r,
        const __bf16* __restrict__ k1r, const __bf16* __restrict__ k2r,
        const float* __restrict__ lamp, __bf16* __restrict__ P) {
    __shared__ float ss[16][RR];
    __shared__ float red[16][8];
    __shared__ float rowv[16];

    int bh   = blockIdx.y;
    int q0   = blockIdx.x << 4;
    int lane = threadIdx.x & 31;
    int wv   = threadIdx.x >> 5;
    int row  = threadIdx.x >> 3;   // 0..15
    int p    = threadIdx.x & 7;
    float lam = *lamp;

    for (int s = 0; s < 2; ++s) {
        const __bf16* qp = ((s == 0) ? q1r : q2r) + ((long)bh * RR + q0) * DD;
        const __bf16* kp = ((s == 0) ? k1r : k2r) + (long)bh * RR * DD;

        v16bf a0 = load_afrag(qp, DD, lane & 15, 0, lane);
        v16bf a1 = load_afrag(qp, DD, lane & 15, 32, lane);

        for (int t = 0; t < 8; ++t) {               // wave handles 8 key tiles
            int kt  = wv * 8 + t;
            int key = (kt << 4) + (lane & 15);
            v16bf b0 = load_bfrag(kp, DD, key, 0, lane);
            v16bf b1 = load_bfrag(kp, DD, key, 32, lane);
            v8f acc = {};
            acc = wmma_bf16(a0, b0, acc);
            acc = wmma_bf16(a1, b1, acc);
            int rb  = (lane >> 4) << 3;
            int col = (kt << 4) + (lane & 15);
#pragma unroll
            for (int j = 0; j < 8; ++j)
                ss[rb + j][col] = acc[j] * 0.125f;  // d^-0.5 = 1/8
        }
        __syncthreads();

        // row max
        float m = -1e30f;
        for (int k = p; k < RR; k += 8) m = fmaxf(m, ss[row][k]);
        red[row][p] = m;
        __syncthreads();
        if (p == 0) {
            float a = red[row][0];
#pragma unroll
            for (int u = 1; u < 8; ++u) a = fmaxf(a, red[row][u]);
            rowv[row] = a;
        }
        __syncthreads();
        m = rowv[row];

        // exp + row sum
        float sum = 0.0f;
        for (int k = p; k < RR; k += 8) {
            float e = __expf(ss[row][k] - m);
            ss[row][k] = e;
            sum += e;
        }
        red[row][p] = sum;
        __syncthreads();
        if (p == 0) {
            float a = 0.0f;
#pragma unroll
            for (int u = 0; u < 8; ++u) a += red[row][u];
            rowv[row] = a;
        }
        __syncthreads();
        float inv = 1.0f / rowv[row];

        __bf16* Pp = P + ((long)bh * RR + q0 + row) * RR;
        if (s == 0) {
            for (int k = p; k < RR; k += 8)
                Pp[k] = (__bf16)(ss[row][k] * inv);
        } else {
            float li = lam * inv;
            for (int k = p; k < RR; k += 8)
                Pp[k] = (__bf16)((float)Pp[k] - ss[row][k] * li);
        }
        __syncthreads();
    }
}

// ---------------------------------------------------------------------------
// head rmsnorm over 2d=128 + *(1-LAMBDA_INIT), cast to bf16
// ---------------------------------------------------------------------------
__global__ __launch_bounds__(128) void headnorm_kernel(const float* __restrict__ diff,
                                                       const float* __restrict__ w,
                                                       __bf16* __restrict__ dn) {
    __shared__ float sh[128];
    int  t   = threadIdx.x;
    long seg = blockIdx.x;          // row*H + h
    long row = seg >> 3;
    int  h   = (int)(seg & 7);
    const float* pp = diff + row * CC + h * D2;
    float v = pp[t];
    sh[t] = v * v;
    __syncthreads();
    for (int st = 64; st > 0; st >>= 1) {
        if (t < st) sh[t] += sh[t + st];
        __syncthreads();
    }
    float rms = sqrtf(sh[0] * (1.0f / (float)D2) + 1e-6f);
    dn[row * CC + h * D2 + t] = (__bf16)(v / rms * w[t] * 0.2f);
}

// ---------------------------------------------------------------------------
// linterp upsample R=512 -> L=4096 (reads lowrank region of d_out)
// ---------------------------------------------------------------------------
__global__ void upsample_kernel(const float* __restrict__ lr, float* __restrict__ full) {
    long idx = (long)blockIdx.x * blockDim.x + threadIdx.x;
    if (idx >= (long)BB * LL * CC) return;
    int  c  = idx & (CC - 1);
    long bl = idx >> 10;
    int  l  = bl & (LL - 1);
    int  b  = (int)(bl >> 12);
    float coord = (l + 0.5f) * ((float)RR / (float)LL) - 0.5f;
    coord = fminf(fmaxf(coord, 0.0f), (float)(RR - 1));
    int lo = (int)floorf(coord);
    int hi = lo + 1 < RR ? lo + 1 : RR - 1;
    float w = coord - (float)lo;
    const float* pp = lr + (long)b * RR * CC + c;
    full[idx] = pp[(long)lo * CC] * (1.0f - w) + pp[(long)hi * CC] * w;
}

// ---------------------------------------------------------------------------
// host launcher
// ---------------------------------------------------------------------------
extern "C" void kernel_launch(void* const* d_in, const int* in_sizes, int n_in,
                              void* d_out, int out_size, void* d_ws, size_t ws_size,
                              hipStream_t stream) {
    const float* x   = (const float*)d_in[0];
    const float* Wq  = (const float*)d_in[1];
    const float* Wk  = (const float*)d_in[2];
    const float* Wv  = (const float*)d_in[3];
    const float* Wo  = (const float*)d_in[4];
    const float* qnw = (const float*)d_in[5];
    const float* knw = (const float*)d_in[6];
    const float* hnw = (const float*)d_in[7];
    const float* lq1 = (const float*)d_in[8];
    const float* lk1 = (const float*)d_in[9];
    const float* lq2 = (const float*)d_in[10];
    const float* lk2 = (const float*)d_in[11];

    float* outFull = (float*)d_out;                       // (B,L,C)
    float* outLow  = outFull + (long)BB * LL * CC;        // (B,R,C)

    char*  ws  = (char*)d_ws;
    size_t off = 0;
    auto give = [&](size_t bytes) -> char* {
        char* pp = ws + off;
        off = (off + bytes + 255) & ~(size_t)255;
        return pp;
    };
    __bf16* xd   = (__bf16*)give((size_t)MT * CC * 2);
    __bf16* Wqb  = (__bf16*)give((size_t)CC * CC * 2);
    __bf16* Wkb  = (__bf16*)give((size_t)CC * CC * 2);
    __bf16* Wvb  = (__bf16*)give((size_t)CC * CC * 2);
    __bf16* Wob  = (__bf16*)give((size_t)CC * CC * 2);
    float*  Qf   = (float*)give((size_t)MT * CC * 4);
    float*  Kf   = (float*)give((size_t)MT * CC * 4);
    float*  Vf   = (float*)give((size_t)MT * CC * 4);
    __bf16* q1r  = (__bf16*)give((size_t)BB * HH * RR * DD * 2);
    __bf16* q2r  = (__bf16*)give((size_t)BB * HH * RR * DD * 2);
    __bf16* k1r  = (__bf16*)give((size_t)BB * HH * RR * DD * 2);
    __bf16* k2r  = (__bf16*)give((size_t)BB * HH * RR * DD * 2);
    __bf16* Vt   = (__bf16*)give((size_t)BB * HH * D2 * RR * 2);
    __bf16* P    = (__bf16*)give((size_t)BB * HH * RR * RR * 2);
    float*  diff = (float*)give((size_t)MT * CC * 4);
    __bf16* dn   = (__bf16*)give((size_t)MT * CC * 2);
    float*  lam  = (float*)give(256);

    const long WN = (long)CC * CC; // 1M weight elements

    // 1. weight conversion + input downsample + lambda
    f2bf_kernel<<<(WN + 255) / 256, 256, 0, stream>>>(Wq, Wqb, WN);
    f2bf_kernel<<<(WN + 255) / 256, 256, 0, stream>>>(Wk, Wkb, WN);
    f2bf_kernel<<<(WN + 255) / 256, 256, 0, stream>>>(Wv, Wvb, WN);
    f2bf_kernel<<<(WN + 255) / 256, 256, 0, stream>>>(Wo, Wob, WN);
    downsample_kernel<<<((long)MT * CC + 255) / 256, 256, 0, stream>>>(x, xd);
    lambda_kernel<<<1, 64, 0, stream>>>(lq1, lk1, lq2, lk2, lam);

    // 2. QKV projections: (2048x1024) = xd @ W^T   (TDM-staged W tiles)
    gemm_bf16_kernel<0, 1><<<dim3(MT / 64, CC / 64, 1), 128, 0, stream>>>(
        xd, Wqb, Qf, MT, CC, CC, 0, 0, 1, 0, 0, CC);
    gemm_bf16_kernel<0, 1><<<dim3(MT / 64, CC / 64, 1), 128, 0, stream>>>(
        xd, Wkb, Kf, MT, CC, CC, 0, 0, 1, 0, 0, CC);
    gemm_bf16_kernel<0, 1><<<dim3(MT / 64, CC / 64, 1), 128, 0, stream>>>(
        xd, Wvb, Vf, MT, CC, CC, 0, 0, 1, 0, 0, CC);

    // 3. rmsnorm + rope (Q and K), V rearrange/transpose
    qknorm_rope_kernel<<<MT * HH * 2, 64, 0, stream>>>(Qf, q1r, q2r, qnw);
    qknorm_rope_kernel<<<MT * HH * 2, 64, 0, stream>>>(Kf, k1r, k2r, knw);
    vtrans_kernel<<<((long)BB * HH * D2 * RR + 255) / 256, 256, 0, stream>>>(Vf, Vt);

    // 4. scores + dual softmax + combine -> P (bf16)
    attn_kernel<<<dim3(RR / 16, BB * HH), 128, 0, stream>>>(q1r, q2r, k1r, k2r, lam, P);

    // 5. diff = P @ V  (batched over 32 (b,h) pairs), async-copy staged
    gemm_bf16_kernel<0, 0><<<dim3(RR / 64, D2 / 64, BB * HH), 128, 0, stream>>>(
        P, Vt, diff, RR, D2, RR,
        (long)RR * RR, (long)D2 * RR,
        HH, (long)D2, (long)RR * CC, CC);

    // 6. head rmsnorm * (1-lambda_init) -> bf16
    headnorm_kernel<<<MT * HH, 128, 0, stream>>>(diff, hnw, dn);

    // 7. lowrank_out = silu(dn @ Wo^T)  -> tail of d_out   (TDM-staged)
    gemm_bf16_kernel<1, 1><<<dim3(MT / 64, CC / 64, 1), 128, 0, stream>>>(
        dn, Wob, outLow, MT, CC, CC, 0, 0, 1, 0, 0, CC);

    // 8. full_out = linterp(lowrank_out, L) -> head of d_out
    upsample_kernel<<<((long)BB * LL * CC + 255) / 256, 256, 0, stream>>>(outLow, outFull);
}
